// ElectraSelfAttention_38534446579839
// MI455X (gfx1250) — matrix operand order repaired and checked by reading, not compiled, not run on hardware
//
#include <hip/hip_runtime.h>

typedef __attribute__((ext_vector_type(16))) __bf16 v16bf;
typedef __attribute__((ext_vector_type(8)))  __bf16 v8bf;
typedef __attribute__((ext_vector_type(8)))  float  v8f;
typedef __attribute__((ext_vector_type(4)))  int    v4i;

#define WMMA_BF16(A, Bm, C) \
  __builtin_amdgcn_wmma_f32_16x16x32_bf16(false, (A), false, (Bm), (short)0, (C), false, false)

#if __has_builtin(__builtin_amdgcn_global_load_async_to_lds_b128)
#define USE_ASYNC_B 1
typedef __attribute__((address_space(1))) v4i gv4i;   // global int4
typedef __attribute__((address_space(3))) v4i lv4i;   // LDS int4
#else
#define USE_ASYNC_B 0
#endif

__device__ __forceinline__ void wait_asynccnt0() {
#if __has_builtin(__builtin_amdgcn_s_wait_asynccnt)
  __builtin_amdgcn_s_wait_asynccnt(0);
#else
  asm volatile("s_wait_asynccnt 0x0" ::: "memory");
#endif
}

constexpr int H = 12, DH = 64, D = 768;
constexpr int B = 32, S = 512, G = 16, NCTX = 8, RB = 4;
constexpr int NROWS = B * S + RB * G;        // 16448 projection rows
constexpr float SCALE = 0.125f;              // 1/sqrt(DH)

// ---------------------------------------------------------------------------
// Kernel 1: convert Wq/Wk/Wv (f32, [k][n]) -> bf16 transposed WT[z][n][k]
// ---------------------------------------------------------------------------
__global__ __launch_bounds__(256)
void convert_w_kernel(const float* __restrict__ Wq, const float* __restrict__ Wk,
                      const float* __restrict__ Wv, __bf16* __restrict__ wt_all) {
  size_t i = (size_t)blockIdx.x * 256 + threadIdx.x;   // 3*D*D total
  int z   = (int)(i / (D * D));
  int rem = (int)(i % (D * D));
  int n = rem / D;
  int k = rem % D;
  const float* W = (z == 0) ? Wq : (z == 1) ? Wk : Wv;
  wt_all[i] = (__bf16)W[(size_t)k * D + n];
}

// ---------------------------------------------------------------------------
// Kernel 2: fused QKV projection GEMM.
//   rows  = [hidden_states (B*S) ; global_tokens (RB*G)]
//   z = 0/1/2 -> Q/K/V, using WT[z]. Q is pre-scaled by 1/sqrt(DH).
//   Q,K stored bf16 as [b][h][s][dh]; V stored transposed [b][h][dh][s].
// Workgroup tile: 128 rows x 64 cols, 8 waves, each wave 16 rows x 64 cols.
// B-panel staging uses GLOBAL_LOAD_ASYNC_TO_LDS (ASYNCcnt) when available,
// overlapped with the A-panel f32->bf16 conversion.
// ---------------------------------------------------------------------------
__global__ __launch_bounds__(256)
void qkv_gemm_kernel(const float* __restrict__ hs, const float* __restrict__ gte,
                     const __bf16* __restrict__ wt_all,
                     const float* __restrict__ bq, const float* __restrict__ bk,
                     const float* __restrict__ bv,
                     __bf16* __restrict__ Qb, __bf16* __restrict__ Kb,
                     __bf16* __restrict__ Vt,
                     __bf16* __restrict__ gQ, __bf16* __restrict__ gK,
                     __bf16* __restrict__ gVt) {
  constexpr int LDA = 40;   // bf16 elems; 80B row stride keeps 16B alignment
  constexpr int LDB = 40;
  __shared__ __bf16 As[128 * LDA];
  __shared__ __bf16 Bs[64 * LDB];

  const int z      = blockIdx.z;
  const int rowBlk = blockIdx.x * 128;
  const int colBlk = blockIdx.y * 64;
  const __bf16* wt   = wt_all + (size_t)z * D * D;
  const float*  bias = (z == 0) ? bq : (z == 1) ? bk : bv;

  const int tid  = threadIdx.x;
  const int lane = tid & 31;
  const int w    = tid >> 5;        // wave 0..7
  const int l15  = lane & 15;
  const bool hi  = lane >= 16;

  v8f acc[4] = {};

  for (int kk = 0; kk < D; kk += 32) {
    // --- stage B block [64 cols x 32 k] from WT (bf16, k-contiguous) ---
    {
      int n  = tid >> 2;
      int k0 = (tid & 3) * 8;
      const __bf16* src = wt + (size_t)(colBlk + n) * D + kk + k0;
      __bf16* dst = &Bs[n * LDB + k0];
#if USE_ASYNC_B
      __builtin_amdgcn_global_load_async_to_lds_b128((gv4i*)src, (lv4i*)dst,
                                                     0, 0);
#else
      *(v8bf*)dst = *(const v8bf*)src;
#endif
    }
    // --- stage A block [128 x 32] (f32 -> bf16), overlaps async B copy ---
    {
      int r  = tid >> 1;
      int k0 = (tid & 1) * 16;
      int rg = rowBlk + r;
      float vals[16];
      if (rg < B * S) {
        const float* p = hs + (size_t)rg * D + kk + k0;
#pragma unroll
        for (int i = 0; i < 16; ++i) vals[i] = p[i];
      } else if (rg < NROWS) {
        const float* p = gte + (size_t)(rg - B * S) * D + kk + k0;
#pragma unroll
        for (int i = 0; i < 16; ++i) vals[i] = p[i];
      } else {
#pragma unroll
        for (int i = 0; i < 16; ++i) vals[i] = 0.f;
      }
      __bf16* dst = &As[r * LDA + k0];
#pragma unroll
      for (int i = 0; i < 16; ++i) dst[i] = (__bf16)vals[i];
    }
#if USE_ASYNC_B
    wait_asynccnt0();
#endif
    __syncthreads();

    // --- fragments + WMMA ---
    {
      int arow = w * 16 + l15;
      int c    = hi ? 8 : 0;
      v8bf alo = *(const v8bf*)(&As[arow * LDA + c]);
      v8bf ahi = *(const v8bf*)(&As[arow * LDA + 16 + c]);
      v16bf af = __builtin_shufflevector(alo, ahi, 0, 1, 2, 3, 4, 5, 6, 7,
                                         8, 9, 10, 11, 12, 13, 14, 15);
      int koff = hi ? 8 : 0;
#pragma unroll
      for (int t = 0; t < 4; ++t) {
        int bn  = t * 16 + l15;
        v8bf b0 = *(const v8bf*)(&Bs[bn * LDB + koff]);
        v8bf b1 = *(const v8bf*)(&Bs[bn * LDB + koff + 16]);
        v16bf bf = __builtin_shufflevector(b0, b1, 0, 1, 2, 3, 4, 5, 6, 7,
                                           8, 9, 10, 11, 12, 13, 14, 15);
        acc[t] = WMMA_BF16(af, bf, acc[t]);
      }
    }
    __syncthreads();
  }

  // --- store C (bias add, Q scaling, layout dispatch) ---
#pragma unroll
  for (int t = 0; t < 4; ++t) {
    int col = colBlk + t * 16 + l15;
    float bval = bias[col];
    int hh = col >> 6;
    int dh = col & 63;
#pragma unroll
    for (int r = 0; r < 8; ++r) {
      int m  = (hi ? 8 : 0) + r;
      int rg = rowBlk + w * 16 + m;
      if (rg >= NROWS) continue;
      float v = acc[t][r] + bval;
      if (z == 0) v *= SCALE;
      __bf16 bvv = (__bf16)v;
      if (rg < B * S) {
        int b = rg / S, s = rg % S;
        if (z == 2)      Vt[((size_t)(b * H + hh) * DH + dh) * S + s]  = bvv;
        else if (z == 0) Qb[((size_t)(b * H + hh) * S + s) * DH + dh]  = bvv;
        else             Kb[((size_t)(b * H + hh) * S + s) * DH + dh]  = bvv;
      } else {
        int gr = rg - B * S;
        int rb = gr >> 4, g = gr & 15;
        if (z == 2)      gVt[((size_t)(rb * H + hh) * DH + dh) * G + g] = bvv;
        else if (z == 0) gQ[((size_t)(rb * H + hh) * G + g) * DH + dh]  = bvv;
        else             gK[((size_t)(rb * H + hh) * G + g) * DH + dh]  = bvv;
      }
    }
  }
}

// ---------------------------------------------------------------------------
// Kernel 3: flash attention.
//   MODE 0: local  — each (b,h): 512 queries over keys = [16 global | 512 own]
//   MODE 1: global — each (rb,h): 16 g-queries over keys = [16 global | 4096 ctx]
// One wave = one 16-row query tile; keys processed 32 at a time with online
// softmax.  Score-tile -> P-fragment transpose via per-wave LDS scratch.
// ---------------------------------------------------------------------------
template <int MODE>
__global__ __launch_bounds__(128)
void attn_kernel(const __bf16* __restrict__ Qb, const __bf16* __restrict__ Kb,
                 const __bf16* __restrict__ Vt, const __bf16* __restrict__ gQ,
                 const __bf16* __restrict__ gK, const __bf16* __restrict__ gVt,
                 const float* __restrict__ mask,
                 float* __restrict__ outCtx, float* __restrict__ outG) {
  __shared__ float sc_all[4][16][33];
  const int lane = threadIdx.x & 31;
  const int w    = threadIdx.x >> 5;
  float (*sc)[33] = sc_all[w];
  const int l15 = lane & 15;
  const bool hi = lane >= 16;

  const int h  = blockIdx.y;
  const int bz = blockIdx.z;                       // b (MODE 0) / rb (MODE 1)
  const int rb = (MODE == 0) ? (bz / NCTX) : bz;
  const int nkeys = (MODE == 0) ? (G + S) : (G + NCTX * S);
  const int nloc  = nkeys - G;

  // ---- Q fragments (dh = K-dim, two 32-wide chunks) ----
  const __bf16* qptr;
  if (MODE == 0) {
    int qrow = blockIdx.x * 64 + w * 16 + l15;
    qptr = Qb + ((size_t)(bz * H + h) * S + qrow) * DH;
  } else {
    qptr = gQ + ((size_t)(rb * H + h) * G + l15) * DH;
  }
  const int c = hi ? 8 : 0;
  v16bf qfr[2];
#pragma unroll
  for (int half = 0; half < 2; ++half) {
    v8bf lo = *(const v8bf*)(qptr + half * 32 + c);
    v8bf hi8 = *(const v8bf*)(qptr + half * 32 + 16 + c);
    qfr[half] = __builtin_shufflevector(lo, hi8, 0, 1, 2, 3, 4, 5, 6, 7,
                                        8, 9, 10, 11, 12, 13, 14, 15);
  }

  float runM = -3.0e38f, runS = 0.f;
  v8f o[4] = {};

  for (int kb = 0; kb < nkeys; kb += 32) {
    // ---- scores: two 16-key tiles, K-dim = DH via 2 chained WMMAs ----
#pragma unroll
    for (int t = 0; t < 2; ++t) {
      int key  = kb + t * 16 + l15;
      int keyc = key < nkeys ? key : nkeys - 1;     // clamp (masked later)
      const __bf16* kr;
      if (keyc < G) {
        kr = gK + ((size_t)(rb * H + h) * G + keyc) * DH;
      } else {
        int tt = keyc - G;
        int bb = (MODE == 0) ? bz : (rb * NCTX + tt / S);
        int ss = (MODE == 0) ? tt : (tt % S);
        kr = Kb + ((size_t)(bb * H + h) * S + ss) * DH;
      }
      v8f scacc = {};
#pragma unroll
      for (int half = 0; half < 2; ++half) {
        int off = half * 32 + (hi ? 8 : 0);
        v8bf lo = *(const v8bf*)(kr + off);
        v8bf hi8 = *(const v8bf*)(kr + off + 16);
        v16bf kf = __builtin_shufflevector(lo, hi8, 0, 1, 2, 3, 4, 5, 6, 7,
                                           8, 9, 10, 11, 12, 13, 14, 15);
        scacc = WMMA_BF16(qfr[half], kf, scacc);
      }
      // additive mask for this lane's key column
      float mval;
      if (key >= nkeys)      mval = -1.0e30f;
      else if (key < G)      mval = 0.f;
      else if (MODE == 0)    mval = mask[bz * S + (key - G)];
      else                   mval = mask[rb * NCTX * S + (key - G)];
#pragma unroll
      for (int r = 0; r < 8; ++r) {
        int m = (hi ? 8 : 0) + r;
        sc[m][t * 16 + l15] = scacc[r] + mval;
      }
    }

    // ---- online softmax over this 32-key panel (row = l15) ----
    float mx = runM;
#pragma unroll
    for (int j = 0; j < 32; ++j) mx = fmaxf(mx, sc[l15][j]);
    float lsum = 0.f;
#pragma unroll
    for (int j = 0; j < 32; ++j) {
      float p = __expf(sc[l15][j] - mx);
      lsum += p;
      sc[l15][j] = p;
    }
    float fac = __expf(runM - mx);
    runS = runS * fac + lsum;
    runM = mx;

    // ---- rescale O accumulators by per-row factor (ds_bpermute bcast) ----
#pragma unroll
    for (int r = 0; r < 8; ++r) {
      int m = (hi ? 8 : 0) + r;
      float fm = __shfl(fac, m, 32);
#pragma unroll
      for (int t = 0; t < 4; ++t) o[t][r] *= fm;
    }

    // ---- P fragment (A-layout) from LDS ----
    v16bf pfr = {};
#pragma unroll
    for (int e = 0; e < 8; ++e) pfr[e] = (__bf16)sc[l15][2 * c + e];
#pragma unroll
    for (int e = 0; e < 8; ++e) pfr[8 + e] = (__bf16)sc[l15][2 * c + 8 + e];

    // ---- O += P x V  (V^T is key-contiguous per dh) ----
    int kg = kb + (hi ? 16 : 0);                 // 16-key group, homogeneous
#pragma unroll
    for (int t = 0; t < 4; ++t) {
      int dh = t * 16 + l15;
      const __bf16* vp;
      if (kg == 0) {
        vp = gVt + ((size_t)(rb * H + h) * DH + dh) * G;
      } else {
        int t0 = kg - G;
        if (t0 > nloc - 16) t0 = nloc - 16;      // clamp invalid tail group
        int bb = (MODE == 0) ? bz : (rb * NCTX + t0 / S);
        int s0 = (MODE == 0) ? t0 : (t0 % S);
        vp = Vt + ((size_t)(bb * H + h) * DH + dh) * S + s0;
      }
      v8bf v0 = *(const v8bf*)(vp);
      v8bf v1 = *(const v8bf*)(vp + 8);
      v16bf vf = __builtin_shufflevector(v0, v1, 0, 1, 2, 3, 4, 5, 6, 7,
                                         8, 9, 10, 11, 12, 13, 14, 15);
      o[t] = WMMA_BF16(pfr, vf, o[t]);
    }
  }

  // ---- normalize and store (f32 outputs) ----
  float inv = 1.0f / runS;
#pragma unroll
  for (int r = 0; r < 8; ++r) {
    int m = (hi ? 8 : 0) + r;
    float im = __shfl(inv, m, 32);
#pragma unroll
    for (int t = 0; t < 4; ++t) {
      float val = o[t][r] * im;
      int dh = t * 16 + l15;
      if (MODE == 0) {
        int s = blockIdx.x * 64 + w * 16 + m;
        outCtx[((size_t)(bz * S + s)) * D + h * DH + dh] = val;
      } else {
        outG[((size_t)(rb * G + m)) * D + h * DH + dh] = val;
      }
    }
  }
}

// ---------------------------------------------------------------------------
extern "C" void kernel_launch(void* const* d_in, const int* in_sizes, int n_in,
                              void* d_out, int out_size, void* d_ws, size_t ws_size,
                              hipStream_t stream) {
  const float* hs   = (const float*)d_in[0];
  const float* mask = (const float*)d_in[1];
  const float* gte  = (const float*)d_in[2];
  const float* Wq = (const float*)d_in[3];
  const float* bq = (const float*)d_in[4];
  const float* Wk = (const float*)d_in[5];
  const float* bk = (const float*)d_in[6];
  const float* Wv = (const float*)d_in[7];
  const float* bv = (const float*)d_in[8];

  char* ws = (char*)d_ws;
  size_t off = 0;
  __bf16* wt = (__bf16*)(ws + off); off += (size_t)3 * D * D * 2;        // 3.4 MB
  __bf16* Qb = (__bf16*)(ws + off); off += (size_t)B * H * S * DH * 2;   // 24 MB
  __bf16* Kb = (__bf16*)(ws + off); off += (size_t)B * H * S * DH * 2;
  __bf16* Vt = (__bf16*)(ws + off); off += (size_t)B * H * S * DH * 2;
  __bf16* gQ = (__bf16*)(ws + off); off += (size_t)RB * H * G * DH * 2;
  __bf16* gK = (__bf16*)(ws + off); off += (size_t)RB * H * G * DH * 2;
  __bf16* gVt = (__bf16*)(ws + off);

  float* outCtx = (float*)d_out;
  float* outG   = outCtx + (size_t)B * S * D;

  // 1) weight transpose + bf16 convert
  convert_w_kernel<<<(3 * D * D) / 256, 256, 0, stream>>>(Wq, Wk, Wv, wt);

  // 2) fused QKV projection (rows = hidden tokens + global tokens)
  dim3 g1((NROWS + 127) / 128, D / 64, 3);
  qkv_gemm_kernel<<<g1, 256, 0, stream>>>(hs, gte, wt, bq, bk, bv,
                                          Qb, Kb, Vt, gQ, gK, gVt);

  // 3) local attention: (q-tile, h, b)
  dim3 g2(S / 64, H, B);
  attn_kernel<0><<<g2, 128, 0, stream>>>(Qb, Kb, Vt, gQ, gK, gVt, mask,
                                         outCtx, outG);

  // 4) global-token cross attention: (1, h, rb)
  dim3 g3(1, H, RB);
  attn_kernel<1><<<g3, 32, 0, stream>>>(Qb, Kb, Vt, gQ, gK, gVt, mask,
                                        outCtx, outG);
}
